// Block_19799799234573
// MI455X (gfx1250) — matrix operand order repaired and checked by reading, not compiled, not run on hardware
//
#include <hip/hip_runtime.h>
#include <cmath>
#include <cstdint>

// ---------------------------------------------------------------------------
// MI455X (gfx1250) fused transformer block.
// B=8, T=1024, C=1024, H=16, D=64, F=16.
// - All GEMM flops on v_wmma_f32_16x16x32_f16 (f16 in, f32 accumulate).
// - GEMMs double-buffer LDS tiles with GLOBAL_LOAD_ASYNC_TO_LDS_B128
//   (ASYNCcnt) so global->LDS overlaps the WMMA stream.
// - Attention is flash-style (scores never touch HBM).
// Fragment layout assumptions (ISA 05_wmma §7.12.2/§7.12.4):
//   A (16x32 f16): lane = m + 16*half, VGPR v holds K-pair at dword
//                  (v&3) + 4*half + 8*(v>>2) of the 32-f16 row.
//   B (32x16 f16): lane = n + 16*half, element s holds k =
//                  (s&7) + 8*half + 16*(s>>3) (same K order as A).
//   C (16x16 f32): VGPR r holds row (r + 8*half), column = lane&15.
// ---------------------------------------------------------------------------

typedef __attribute__((ext_vector_type(16))) _Float16 v16h;
typedef __attribute__((ext_vector_type(8)))  float    v8f;

union Frag16 { v16h h; uint32_t u[8]; };

__device__ __forceinline__ v16h load_fragA(const uint32_t* row16dw, int half) {
  Frag16 f;
#pragma unroll
  for (int v = 0; v < 8; ++v)
    f.u[v] = row16dw[(v & 3) + (half << 2) + ((v >> 2) << 3)];
  return f.h;
}

__device__ __forceinline__ v16h load_fragB(const _Float16* col, int ld, int half) {
  v16h f;
#pragma unroll
  for (int s = 0; s < 16; ++s)
    f[s] = col[((s & 7) + (half << 3) + ((s >> 3) << 4)) * ld];
  return f;
}

__device__ __forceinline__ v8f wmma_f16(v16h a, v16h b, v8f c) {
  return __builtin_amdgcn_wmma_f32_16x16x32_f16(false, a, false, b, (short)0, c,
                                                false, false);
}

// Async 16-byte global->LDS copy (GLOBAL_LOAD_ASYNC_TO_LDS_B128, GVS mode).
// Generic pointers to LDS carry the LDS byte offset in their low 32 bits.
__device__ __forceinline__ void async_copy16(void* lds, const void* gbase,
                                             int gbyteoff) {
  uint32_t l = (uint32_t)(uintptr_t)lds;
  asm volatile("global_load_async_to_lds_b128 %0, %1, %2 offset:0"
               :: "v"(l), "v"(gbyteoff), "s"(gbase)
               : "memory");
}

__device__ __forceinline__ void wait_async0() {
#if __has_builtin(__builtin_amdgcn_s_wait_asynccnt)
  __builtin_amdgcn_s_wait_asynccnt(0);
#else
  asm volatile("s_wait_asynccnt 0x0" ::: "memory");
#endif
}

// ---------------------------------------------------------------------------
// f32 -> f16 convert (weights), grid-stride
// ---------------------------------------------------------------------------
__global__ void cvt_f32_f16(const float* __restrict__ in,
                            _Float16* __restrict__ out, int n) {
  for (int i = blockIdx.x * blockDim.x + threadIdx.x; i < n;
       i += gridDim.x * blockDim.x)
    out[i] = (_Float16)in[i];
}

// ---------------------------------------------------------------------------
// LayerNorm over rows of [rows, 1024], f32 in -> f16 out
// ---------------------------------------------------------------------------
__global__ __launch_bounds__(256, 1)
void ln_rows(const float* __restrict__ x, const float* __restrict__ g,
             const float* __restrict__ bta, _Float16* __restrict__ out) {
  const int C = 1024;
  const int row = blockIdx.x;
  const float* xr = x + (size_t)row * C;
  const int tid = threadIdx.x, lane = tid & 31, wave = tid >> 5;
  float v[4];
  float s = 0.f, s2 = 0.f;
#pragma unroll
  for (int j = 0; j < 4; ++j) {
    v[j] = xr[tid + j * 256];
    s += v[j];
    s2 += v[j] * v[j];
  }
#pragma unroll
  for (int m = 16; m >= 1; m >>= 1) {
    s += __shfl_xor(s, m, 32);
    s2 += __shfl_xor(s2, m, 32);
  }
  __shared__ float rs[8], rs2[8], stat[2];
  if (lane == 0) { rs[wave] = s; rs2[wave] = s2; }
  __syncthreads();
  if (tid == 0) {
    float a = 0.f, b2 = 0.f;
    for (int w = 0; w < 8; ++w) { a += rs[w]; b2 += rs2[w]; }
    float mu = a * (1.f / C);
    float var = b2 * (1.f / C) - mu * mu;
    stat[0] = mu;
    stat[1] = rsqrtf(var + 1e-5f);
  }
  __syncthreads();
  const float mu = stat[0], rstd = stat[1];
#pragma unroll
  for (int j = 0; j < 4; ++j) {
    int i = tid + j * 256;
    out[(size_t)row * C + i] = (_Float16)((v[j] - mu) * rstd * g[i] + bta[i]);
  }
}

// ---------------------------------------------------------------------------
// pos[h,t,s] = sum_f bias[t,s,f] * W_bias[f,h] + b_bias[h]   (f32, 64 MB)
// ---------------------------------------------------------------------------
__global__ __launch_bounds__(256, 1)
void pos_bias(const float* __restrict__ bias, const float* __restrict__ Wb,
              const float* __restrict__ bb, float* __restrict__ pos) {
  __shared__ float w[16][16];
  __shared__ float bbs[16];
  w[threadIdx.x >> 4][threadIdx.x & 15] = Wb[threadIdx.x];
  if (threadIdx.x < 16) bbs[threadIdx.x] = bb[threadIdx.x];
  __syncthreads();
  const size_t idx = (size_t)blockIdx.x * 256 + threadIdx.x;  // t*1024 + s
  const float* br = bias + (idx << 4);
  float f[16];
#pragma unroll
  for (int i = 0; i < 16; ++i) f[i] = br[i];
#pragma unroll
  for (int h = 0; h < 16; ++h) {
    float acc = bbs[h];
#pragma unroll
    for (int i = 0; i < 16; ++i) acc += f[i] * w[i][h];
    pos[((size_t)h << 20) + idx] = acc;
  }
}

// ---------------------------------------------------------------------------
// qkv f32 [B,T,3C] -> Q,K,V f16 [B,H,T,D]
// ---------------------------------------------------------------------------
__global__ __launch_bounds__(256, 1)
void qkv_split(const float* __restrict__ qkv, _Float16* __restrict__ Q,
               _Float16* __restrict__ Kk, _Float16* __restrict__ V) {
  const size_t i = (size_t)blockIdx.x * 256 + threadIdx.x;  // over B*T*C = 8M
  const int d = (int)(i & 63);
  const int h = (int)((i >> 6) & 15);
  const size_t bt = i >> 10;
  const int t = (int)(bt & 1023);
  const int b = (int)(bt >> 10);
  const size_t src = bt * 3072;
  const size_t dst = (((size_t)b * 16 + h) * 1024 + t) * 64 + d;
  const int c = h * 64 + d;
  Q[dst]  = (_Float16)qkv[src + c];
  Kk[dst] = (_Float16)qkv[src + 1024 + c];
  V[dst]  = (_Float16)qkv[src + 2048 + c];
}

// ---------------------------------------------------------------------------
// Generic f16 GEMM: out[M,N] = A[M,K] @ B[K,N] + bias (+ epilogue)
// MODE 0: f32 store; MODE 1: f32 store + residual; MODE 2: GELU -> f16 store
// Block tile 128x128, BK=32, 8 waves (4 along M x 2 along N), each wave
// computes 32x64 via 2x4 16x16 WMMA accumulators. LDS double-buffered with
// async global->LDS b128 copies overlapping the WMMA stream.
// ---------------------------------------------------------------------------
template <int MODE>
__global__ __launch_bounds__(256, 1)
void gemm_f16k(const _Float16* __restrict__ A, const _Float16* __restrict__ B,
               const float* __restrict__ bias, const float* __restrict__ res,
               void* __restrict__ outv, int M, int N, int K) {
  __shared__ __align__(16) _Float16 sA[2][128][40];   // [m][k], pad 32->40
  __shared__ __align__(16) _Float16 sB[2][32][136];   // [k][n], pad 128->136
  const int tid = threadIdx.x, lane = tid & 31, wave = tid >> 5;
  const int wm = wave >> 1, wn = wave & 1;
  const int half = lane >> 4, l16 = lane & 15;
  const int bm = blockIdx.y * 128, bn = blockIdx.x * 128;
  v8f acc[2][4] = {};

  // Stage one 128x32 A tile + 32x128 B tile into buffer `buf` (async).
  auto stage = [&](int buf, int k0) {
#pragma unroll
    for (int j = 0; j < 2; ++j) {  // A: 512 16B chunks / 256 threads
      const int id = tid * 2 + j;
      const int m = id >> 2, c = id & 3;  // 4 chunks per 32-f16 row
      async_copy16(&sA[buf][m][c * 8], A, ((bm + m) * K + k0 + c * 8) * 2);
    }
#pragma unroll
    for (int j = 0; j < 2; ++j) {  // B: 512 16B chunks / 256 threads
      const int id = tid * 2 + j;
      const int k = id >> 4, c = id & 15;  // 16 chunks per 128-f16 row
      async_copy16(&sB[buf][k][c * 8], B, ((k0 + k) * N + bn + c * 8) * 2);
    }
  };

  stage(0, 0);
  wait_async0();
  __syncthreads();

  const int ksteps = K >> 5;
  for (int ks = 0; ks < ksteps; ++ks) {
    const int buf = ks & 1;
    if (ks + 1 < ksteps) stage(buf ^ 1, (ks + 1) << 5);

    v16h af[2], bf[4];
#pragma unroll
    for (int fm = 0; fm < 2; ++fm)
      af[fm] = load_fragA((const uint32_t*)&sA[buf][wm * 32 + fm * 16 + l16][0],
                          half);
#pragma unroll
    for (int fn = 0; fn < 4; ++fn)
      bf[fn] = load_fragB(&sB[buf][0][wn * 64 + fn * 16 + l16], 136, half);
#pragma unroll
    for (int fm = 0; fm < 2; ++fm)
#pragma unroll
      for (int fn = 0; fn < 4; ++fn)
        acc[fm][fn] = wmma_f16(af[fm], bf[fn], acc[fm][fn]);

    wait_async0();   // next tile fully landed in LDS
    __syncthreads(); // all waves done reading current tile
  }

  float* outF = (float*)outv;
  _Float16* outH = (_Float16*)outv;
#pragma unroll
  for (int fm = 0; fm < 2; ++fm)
#pragma unroll
    for (int fn = 0; fn < 4; ++fn)
#pragma unroll
      for (int r = 0; r < 8; ++r) {
        const int grow = bm + wm * 32 + fm * 16 + r + half * 8;
        const int gcol = bn + wn * 64 + fn * 16 + l16;
        float v = acc[fm][fn][r] + bias[gcol];
        const size_t o = (size_t)grow * N + gcol;
        if (MODE == 0) {
          outF[o] = v;
        } else if (MODE == 1) {
          outF[o] = v + res[o];
        } else {
          const float c0 = 0.7978845608028654f;
          float g = 0.5f * v * (1.f + tanhf(c0 * (v + 0.044715f * v * v * v)));
          outH[o] = (_Float16)g;
        }
      }
}

// ---------------------------------------------------------------------------
// Flash attention: one block per (b, h, 64-row q tile); 4 waves x 16 q rows.
// Per 32-key step: S = Q Kt (4 WMMA) -> +pos, online softmax -> P via LDS
// relayout -> O += P V (4 WMMA). V tiles stream in via async-to-LDS.
// ---------------------------------------------------------------------------
__global__ __launch_bounds__(128, 1)
void flash_attn(const _Float16* __restrict__ Qm, const _Float16* __restrict__ Km,
                const _Float16* __restrict__ Vm, const float* __restrict__ pos,
                _Float16* __restrict__ Y) {
  __shared__ __align__(16) _Float16 sKT[64][40];     // [d][s]: B-frag lane = s
  __shared__ __align__(16) _Float16 sV[32][72];      // [s][d]: B-frag lane = d
  __shared__ __align__(16) _Float16 sP[4][16][32];   // per-wave P relayout
  const int tid = threadIdx.x, lane = tid & 31, wave = tid >> 5;
  const int half = lane >> 4, l16 = lane & 15;
  const int b = blockIdx.z, h = blockIdx.y, qt = blockIdx.x;
  const size_t head = ((size_t)(b * 16 + h)) * (1024 * 64);
  const int qbase = qt * 64 + wave * 16;
  const _Float16* Vh = Vm + head;

  // Q fragments straight from global (row = contiguous 64 f16).
  const uint32_t* qrow = (const uint32_t*)(Qm + head + (size_t)(qbase + l16) * 64);
  const v16h aq0 = load_fragA(qrow, half);       // K(d)=0..31
  const v16h aq1 = load_fragA(qrow + 16, half);  // K(d)=32..63

  v8f o0 = {}, o1 = {}, o2 = {}, o3 = {};
  float mrow[8], lrow[8];
#pragma unroll
  for (int r = 0; r < 8; ++r) { mrow[r] = -3.0e38f; lrow[r] = 0.f; }
  const float scale = 0.125f;  // 1/sqrt(64)
  const float* posh = pos + ((size_t)h << 20);

  for (int s0 = 0; s0 < 1024; s0 += 32) {
    __syncthreads();  // prior iteration done reading sKT/sV/sP
    // V tile 32x64, row-major: direct async copy (256 chunks / 128 threads).
#pragma unroll
    for (int j = 0; j < 2; ++j) {
      const int id = tid * 2 + j;
      const int sr = id >> 3, c = id & 7;  // 8 chunks per 64-f16 row
      async_copy16(&sV[sr][c * 8], Vh, ((s0 + sr) * 64 + c * 8) * 2);
    }
    // Kt tile: transpose K[s][d] -> sKT[d][s] (f16 scatter).
    for (int e = tid; e < 2048; e += 128) {
      const int sr = e >> 6, d = e & 63;
      sKT[d][sr] = Km[head + (size_t)(s0 + sr) * 64 + d];
    }
    wait_async0();
    __syncthreads();

    v8f S0 = {}, S1 = {};
    S0 = wmma_f16(aq0, load_fragB(&sKT[0][l16], 40, half), S0);
    S0 = wmma_f16(aq1, load_fragB(&sKT[32][l16], 40, half), S0);
    S1 = wmma_f16(aq0, load_fragB(&sKT[0][16 + l16], 40, half), S1);
    S1 = wmma_f16(aq1, load_fragB(&sKT[32][16 + l16], 40, half), S1);

#pragma unroll
    for (int r = 0; r < 8; ++r) {
      const int t = qbase + r + half * 8;
      const float* prow = posh + ((size_t)t << 10) + s0;
      float v0 = S0[r] * scale + prow[l16];
      float v1 = S1[r] * scale + prow[16 + l16];
      float mx = fmaxf(v0, v1);
      mx = fmaxf(mx, __shfl_xor(mx, 1, 32));
      mx = fmaxf(mx, __shfl_xor(mx, 2, 32));
      mx = fmaxf(mx, __shfl_xor(mx, 4, 32));
      mx = fmaxf(mx, __shfl_xor(mx, 8, 32));
      const float mn = fmaxf(mrow[r], mx);
      const float alpha = __expf(mrow[r] - mn);
      mrow[r] = mn;
      const float p0 = __expf(v0 - mn), p1 = __expf(v1 - mn);
      float rsum = p0 + p1;
      rsum += __shfl_xor(rsum, 1, 32);
      rsum += __shfl_xor(rsum, 2, 32);
      rsum += __shfl_xor(rsum, 4, 32);
      rsum += __shfl_xor(rsum, 8, 32);
      lrow[r] = lrow[r] * alpha + rsum;
      o0[r] *= alpha; o1[r] *= alpha; o2[r] *= alpha; o3[r] *= alpha;
      const int rl = r + half * 8;
      sP[wave][rl][l16] = (_Float16)p0;
      sP[wave][rl][16 + l16] = (_Float16)p1;
    }
    __syncthreads();  // order sP writes before fragment readback

    const v16h ap = load_fragA((const uint32_t*)&sP[wave][l16][0], half);
    o0 = wmma_f16(ap, load_fragB(&sV[0][l16], 72, half), o0);
    o1 = wmma_f16(ap, load_fragB(&sV[0][16 + l16], 72, half), o1);
    o2 = wmma_f16(ap, load_fragB(&sV[0][32 + l16], 72, half), o2);
    o3 = wmma_f16(ap, load_fragB(&sV[0][48 + l16], 72, half), o3);
  }

#pragma unroll
  for (int r = 0; r < 8; ++r) {
    const float inv = 1.f / lrow[r];
    const int t = qbase + r + half * 8;
    const size_t base = ((size_t)b * 1024 + t) * 1024 + h * 64;
    Y[base + l16]      = (_Float16)(o0[r] * inv);
    Y[base + 16 + l16] = (_Float16)(o1[r] * inv);
    Y[base + 32 + l16] = (_Float16)(o2[r] * inv);
    Y[base + 48 + l16] = (_Float16)(o3[r] * inv);
  }
}

// ---------------------------------------------------------------------------
// Host orchestration
// ---------------------------------------------------------------------------
extern "C" void kernel_launch(void* const* d_in, const int* in_sizes, int n_in,
                              void* d_out, int out_size, void* d_ws,
                              size_t ws_size, hipStream_t stream) {
  const float* x      = (const float*)d_in[0];
  const float* bias   = (const float*)d_in[1];
  const float* ln1_g  = (const float*)d_in[2];
  const float* ln1_b  = (const float*)d_in[3];
  const float* W_attn = (const float*)d_in[4];
  const float* b_attn = (const float*)d_in[5];
  const float* W_o    = (const float*)d_in[6];
  const float* b_o    = (const float*)d_in[7];
  const float* W_bias = (const float*)d_in[8];
  const float* b_bias = (const float*)d_in[9];
  const float* ln2_g  = (const float*)d_in[10];
  const float* ln2_b  = (const float*)d_in[11];
  const float* W_fc   = (const float*)d_in[12];
  const float* b_fc   = (const float*)d_in[13];
  const float* W_fc2  = (const float*)d_in[14];
  const float* b_fc2  = (const float*)d_in[15];
  float* out = (float*)d_out;

  const int BT = 8 * 1024;  // 8192 rows
  char* ws = (char*)d_ws;
  size_t off = 0;
  auto take = [&](size_t bytes) -> void* {
    void* p = ws + off;
    off += (bytes + 255) & ~(size_t)255;
    return p;
  };
  _Float16* Wattn16 = (_Float16*)take((size_t)1024 * 3072 * 2);
  _Float16* Wo16    = (_Float16*)take((size_t)1024 * 1024 * 2);
  _Float16* Wfc16   = (_Float16*)take((size_t)1024 * 4096 * 2);
  _Float16* Wfc216  = (_Float16*)take((size_t)4096 * 1024 * 2);
  _Float16* hA      = (_Float16*)take((size_t)BT * 1024 * 2);   // ln1/ln2 out
  float*    qkvf    = (float*)take((size_t)BT * 3072 * 4);      // reused: FC1 out
  _Float16* Q16     = (_Float16*)take((size_t)BT * 1024 * 2);
  _Float16* K16     = (_Float16*)take((size_t)BT * 1024 * 2);
  _Float16* V16     = (_Float16*)take((size_t)BT * 1024 * 2);
  float*    posb    = (float*)take((size_t)16 * 1024 * 1024 * 4);
  _Float16* y16     = (_Float16*)take((size_t)BT * 1024 * 2);
  _Float16* h2  = hA;               // alias: hA dead after QKV GEMM
  _Float16* hfc = (_Float16*)qkvf;  // alias: qkv dead after split (64MB <= 96MB)

  cvt_f32_f16<<<2048, 256, 0, stream>>>(W_attn, Wattn16, 1024 * 3072);
  cvt_f32_f16<<<2048, 256, 0, stream>>>(W_o, Wo16, 1024 * 1024);
  cvt_f32_f16<<<2048, 256, 0, stream>>>(W_fc, Wfc16, 1024 * 4096);
  cvt_f32_f16<<<2048, 256, 0, stream>>>(W_fc2, Wfc216, 4096 * 1024);
  ln_rows<<<BT, 256, 0, stream>>>(x, ln1_g, ln1_b, hA);
  pos_bias<<<4096, 256, 0, stream>>>(bias, W_bias, b_bias, posb);
  gemm_f16k<0><<<dim3(24, 64), 256, 0, stream>>>(hA, Wattn16, b_attn, nullptr,
                                                 qkvf, BT, 3072, 1024);
  qkv_split<<<32768, 256, 0, stream>>>(qkvf, Q16, K16, V16);
  flash_attn<<<dim3(16, 16, 8), 128, 0, stream>>>(Q16, K16, V16, posb, y16);
  gemm_f16k<1><<<dim3(8, 64), 256, 0, stream>>>(y16, Wo16, b_o, x, out, BT,
                                                1024, 1024);
  ln_rows<<<BT, 256, 0, stream>>>(out, ln2_g, ln2_b, h2);
  gemm_f16k<2><<<dim3(32, 64), 256, 0, stream>>>(h2, Wfc16, b_fc, nullptr, hfc,
                                                 BT, 4096, 1024);
  gemm_f16k<1><<<dim3(8, 64), 256, 0, stream>>>(hfc, Wfc216, b_fc2, out, out,
                                                BT, 1024, 4096);
  (void)in_sizes; (void)n_in; (void)out_size; (void)ws_size;
}
